// DimeNetBackbone_30932354466265
// MI455X (gfx1250) — matrix operand order repaired and checked by reading, not compiled, not run on hardware
//
#include <hip/hip_runtime.h>

// ---------------------------------------------------------------------------
// DimeNet++ backbone for MI455X (gfx1250), wave32.
// bf16 WMMA 16x16x32 with f32 accumulate; async global->LDS staging.
// ---------------------------------------------------------------------------

typedef __attribute__((ext_vector_type(16))) __bf16 v16bf;
typedef __attribute__((ext_vector_type(8)))  float  v8f;

#define HIDDEN 128
#define PI_F 3.14159265358979f
#define PI_D 3.141592653589793238462643

__device__ __forceinline__ float silu_f(float v) { return v / (1.0f + __expf(-v)); }

// Async copy of 16 bytes global -> LDS on the CDNA5 async engine (ASYNCcnt).
__device__ __forceinline__ void async_g2l_b128(unsigned lds_off, const void* gptr) {
    asm volatile("global_load_async_to_lds_b128 %0, %1, off"
                 :: "v"(lds_off), "v"((unsigned long long)(size_t)gptr)
                 : "memory");
}
__device__ __forceinline__ void wait_async0() {
    asm volatile("s_wait_asynccnt 0x0" ::: "memory");
}

// ---------------------------------------------------------------------------
// Bessel zeros + norms computed on device (reference derives them in numpy).
// ---------------------------------------------------------------------------
__device__ double sph_jn_d(double x, int n) {
    double j0 = sin(x) / x;
    if (n == 0) return j0;
    double j1 = sin(x) / (x * x) - cos(x) / x;
    double jm = j0, jc = j1;
    for (int k = 1; k < n; ++k) { double t = (2.0 * k + 1.0) / x * jc - jm; jm = jc; jc = t; }
    return jc;
}

__global__ void bessel_setup_kernel(float* zeros /*42*/, float* norm /*42*/) {
    if (threadIdx.x != 0 || blockIdx.x != 0) return;
    const int NSPH = 7, NRAD = 6;
    double zer[NSPH][NRAD];
    double points[NRAD + NSPH];
    for (int i = 0; i < NRAD; ++i) zer[0][i] = (i + 1) * PI_D;
    for (int j = 0; j < NRAD + NSPH - 1; ++j) points[j] = (j + 1) * PI_D;
    for (int i = 1; i < NSPH; ++i) {
        int nr = NRAD + NSPH - 1 - i;
        double rac[NRAD + NSPH - 1];
        for (int j = 0; j < nr; ++j) {
            double a = points[j], b = points[j + 1];
            double fa = sph_jn_d(a, i);
            for (int it = 0; it < 80; ++it) {
                double m = 0.5 * (a + b);
                double fm = sph_jn_d(m, i);
                if (fa * fm <= 0.0) { b = m; } else { a = m; fa = fm; }
            }
            rac[j] = 0.5 * (a + b);
        }
        for (int j = 0; j < nr; ++j) points[j] = rac[j];
        for (int j = 0; j < NRAD; ++j) zer[i][j] = points[j];
    }
    for (int l = 0; l < NSPH; ++l)
        for (int r = 0; r < NRAD; ++r) {
            zeros[l * 6 + r] = (float)zer[l][r];
            double jv = sph_jn_d(zer[l][r], l + 1);
            norm[l * 6 + r] = (float)(sqrt(2.0) / fabs(jv));
        }
}

// ---------------------------------------------------------------------------
// Edge geometry: rbf (E x 6) and rbf_s (E x 42), envelope fused in.
// ---------------------------------------------------------------------------
__device__ __forceinline__ float sph_jl_f(float x, int l) {
    float s = sinf(x), c = cosf(x);
    float j0 = s / x;
    if (l == 0) return j0;
    float j1 = s / (x * x) - c / x;
    float jm = j0, jc = j1;
    for (int k = 1; k < l; ++k) { float t = (2.0f * k + 1.0f) / x * jc - jm; jm = jc; jc = t; }
    return jc;
}

__global__ void edge_geom_kernel(const float* __restrict__ pos,
                                 const int* __restrict__ ei, const int* __restrict__ ej,
                                 const float* __restrict__ zeros, const float* __restrict__ norm,
                                 float* __restrict__ rbf, float* __restrict__ rbfs, int E) {
    int e = blockIdx.x * 256 + threadIdx.x;
    if (e >= E) return;
    int i = ei[e], j = ej[e];
    float dx = pos[i * 3 + 0] - pos[j * 3 + 0];
    float dy = pos[i * 3 + 1] - pos[j * 3 + 1];
    float dz = pos[i * 3 + 2] - pos[j * 3 + 2];
    float dist = sqrtf(dx * dx + dy * dy + dz * dz);
    float d = dist / 5.0f;  // CUTOFF
    float env = 0.0f;
    if (d < 1.0f) {
        float x5 = d * d; x5 *= x5; x5 *= d;  // d^5
        env = 1.0f / d - 28.0f * x5 + 48.0f * x5 * d - 21.0f * x5 * d * d;
    }
    for (int r = 0; r < 6; ++r) rbf[(long)e * 6 + r] = env * sinf((r + 1) * PI_F * d);
    for (int l = 0; l < 7; ++l)
        for (int r = 0; r < 6; ++r) {
            float x = zeros[l * 6 + r] * d;
            rbfs[(long)e * 42 + l * 6 + r] = env * norm[l * 6 + r] * sph_jl_f(x, l);
        }
}

// ---------------------------------------------------------------------------
// Triplet geometry: angle -> scaled Legendre cbf (T x 7).
// ---------------------------------------------------------------------------
__global__ void triplet_geom_kernel(const float* __restrict__ pos,
                                    const int* __restrict__ ii, const int* __restrict__ ij,
                                    const int* __restrict__ ik,
                                    float* __restrict__ cbf, int T) {
    int t = blockIdx.x * 256 + threadIdx.x;
    if (t >= T) return;
    int a = ii[t], b = ij[t], c = ik[t];
    float pix = pos[a * 3 + 0], piy = pos[a * 3 + 1], piz = pos[a * 3 + 2];
    float jx = pos[b * 3 + 0] - pix, jy = pos[b * 3 + 1] - piy, jz = pos[b * 3 + 2] - piz;
    float kx = pos[c * 3 + 0] - pix, ky = pos[c * 3 + 1] - piy, kz = pos[c * 3 + 2] - piz;
    float dot = jx * kx + jy * ky + jz * kz;
    float cx = jy * kz - jz * ky, cy = jz * kx - jx * kz, cz = jx * ky - jy * kx;
    float cr = sqrtf(cx * cx + cy * cy + cz * cz);
    float ang = atan2f(cr, dot);
    float ct = cosf(ang);
    const float inv4pi = 0.07957747154594767f;
    float P0 = 1.0f, P1 = ct;
    cbf[(long)t * 7 + 0] = P0 * sqrtf(1.0f * inv4pi);
    cbf[(long)t * 7 + 1] = P1 * sqrtf(3.0f * inv4pi);
    float Pm = P0, Pc = P1;
    for (int l = 1; l < 6; ++l) {
        float Pn = ((2.0f * l + 1.0f) * ct * Pc - (float)l * Pm) / (float)(l + 1);
        cbf[(long)t * 7 + l + 1] = Pn * sqrtf((2.0f * (l + 1) + 1.0f) * inv4pi);
        Pm = Pc; Pc = Pn;
    }
}

// ---------------------------------------------------------------------------
// Small elementwise / gather / pack kernels.
// ---------------------------------------------------------------------------
__global__ void gather_emb_kernel(const float* __restrict__ emb, const int* __restrict__ z,
                                  __bf16* __restrict__ out, int N) {
    int i = blockIdx.x * 256 + threadIdx.x;
    if (i >= N * HIDDEN) return;
    int n = i >> 7, c = i & 127;
    out[i] = (__bf16)emb[(long)z[n] * HIDDEN + c];
}

__global__ void rbf_emb_kernel(const float* __restrict__ rbf, const float* __restrict__ W,
                               const float* __restrict__ bias, __bf16* __restrict__ out, int E) {
    int i = blockIdx.x * 256 + threadIdx.x;
    if (i >= E * HIDDEN) return;
    int e = i >> 7, c = i & 127;
    float acc = bias[c];
    for (int q = 0; q < 6; ++q) acc += rbf[(long)e * 6 + q] * W[q * HIDDEN + c];
    out[i] = (__bf16)silu_f(acc);
}

__global__ void f2b_kernel(const float* __restrict__ in, __bf16* __restrict__ out, int n) {
    int i = blockIdx.x * 256 + threadIdx.x;
    if (i < n) out[i] = (__bf16)in[i];
}

__global__ void add2_kernel(const float* __restrict__ a, const float* __restrict__ b,
                            float* __restrict__ of, __bf16* __restrict__ ob, int n) {
    int i = blockIdx.x * 256 + threadIdx.x;
    if (i >= n) return;
    float v = a[i] + b[i];
    of[i] = v;
    ob[i] = (__bf16)v;
}

// pack W (K x 128 row-major, f32) -> Bt (128 x K, N-major, bf16)
__global__ void pack_weight_kernel(const float* __restrict__ W, __bf16* __restrict__ Bt,
                                   int K, int total) {
    int i = blockIdx.x * 256 + threadIdx.x;
    if (i >= total) return;
    int n = i / K, k = i - n * K;
    Bt[i] = (__bf16)W[(long)k * HIDDEN + n];
}

// t = (rbf @ out_Wrbf[k]) * x, scatter-add into node accumulator
__global__ void outblock_scatter_kernel(const float* __restrict__ rbf, const float* __restrict__ Wrbf,
                                        const float* __restrict__ x, const int* __restrict__ ei,
                                        float* __restrict__ tnode, int E) {
    int i = blockIdx.x * 256 + threadIdx.x;
    if (i >= E * HIDDEN) return;
    int e = i >> 7, c = i & 127;
    float rb = 0.0f;
    for (int q = 0; q < 6; ++q) rb += rbf[(long)e * 6 + q] * Wrbf[q * HIDDEN + c];
    atomicAdd(&tnode[(long)ei[e] * HIDDEN + c], rb * x[i]);
}

// ---------------------------------------------------------------------------
// Generic WMMA GEMM, compile-time fused epilogue.
//   [M x 128] bf16 (optional row gather) @ Bt[128][ldb] (N-major bf16)
// 32 rows per block (two 16-row tiles share each B fragment), 8 waves,
// wave w owns columns [16w,16w+16). A staged via async global->LDS.
// ---------------------------------------------------------------------------
enum {
    GM_CIN = 1, GM_BIAS = 2, GM_ACT = 4, GM_RB = 8,
    GM_RES = 16, GM_OUTF = 32, GM_OUTB = 64
};

template <int MODE>
__global__ __launch_bounds__(256) void gemm_tile_kernel(
    const __bf16* __restrict__ A, const int* __restrict__ rowmap,
    const __bf16* __restrict__ Bt, int ldb, int kOff,
    const float* __restrict__ bias, const float* __restrict__ Cin,
    const float* __restrict__ resid,
    const float* __restrict__ rbfE, const float* __restrict__ Wrbf6,
    float* __restrict__ outF, __bf16* __restrict__ outB, int M) {
    const int K = 128;
    __shared__ __bf16 ldsA[32 * 128];

    int row0 = blockIdx.x * 32;
    unsigned ldsBase = (unsigned)(size_t)&ldsA[0];  // LDS byte offset (low 32 bits)
    // 32 rows x 256B = 8KB staged as 512 async 16B copies (2 per thread).
#pragma unroll
    for (int s = 0; s < 2; ++s) {
        int idx = threadIdx.x + s * 256;
        int r = idx >> 4, ch = idx & 15;
        int grow = row0 + r; if (grow >= M) grow = M - 1;
        int arow = rowmap ? rowmap[grow] : grow;
        async_g2l_b128(ldsBase + (unsigned)idx * 16, A + (long)arow * K + ch * 8);
    }
    wait_async0();
    __syncthreads();

    int lane = threadIdx.x & 31;
    int wave = threadIdx.x >> 5;
    int half = lane >> 4;
    int ln   = lane & 15;
    int col0 = wave * 16;

    v8f acc0 = {}, acc1 = {};
#pragma unroll
    for (int k0 = 0; k0 < K; k0 += 32) {
        v16bf a0, a1, bv;
        const __bf16* Ar0 = &ldsA[ln * K + k0];
        const __bf16* Ar1 = &ldsA[(16 + ln) * K + k0];
        const __bf16* Bc  = Bt + (long)(col0 + ln) * ldb + kOff + k0;
#pragma unroll
        for (int v = 0; v < 8; ++v) {
            int ka = ((v & 4) << 2) + ((v & 3) << 1) + (half << 3);  // A 16-bit layout
            a0[2 * v]     = Ar0[ka];
            a0[2 * v + 1] = Ar0[ka + 1];
            a1[2 * v]     = Ar1[ka];
            a1[2 * v + 1] = Ar1[ka + 1];
            int kb = ka;                                              // B matches A k order
            bv[2 * v]     = Bc[kb];
            bv[2 * v + 1] = Bc[kb + 1];
        }
        acc0 = __builtin_amdgcn_wmma_f32_16x16x32_bf16(false, a0, false, bv,
                                                       (short)0, acc0, false, false);
        acc1 = __builtin_amdgcn_wmma_f32_16x16x32_bf16(false, a1, false, bv,
                                                       (short)0, acc1, false, false);
    }

    int col = col0 + ln;
#pragma unroll
    for (int tgt = 0; tgt < 2; ++tgt) {
        v8f acc = tgt ? acc1 : acc0;
        int rbase = row0 + tgt * 16;
#pragma unroll
        for (int r = 0; r < 8; ++r) {
            int m = r + (half << 3);
            int grow = rbase + m;
            if (grow >= M) continue;
            long o = (long)grow * HIDDEN + col;
            float v = acc[r];
            if (MODE & GM_CIN)  v += Cin[o];
            if (MODE & GM_BIAS) v += bias[col];
            if (MODE & GM_ACT)  v = silu_f(v);
            if (MODE & GM_RB) {
                float rb = 0.0f;
#pragma unroll
                for (int q = 0; q < 6; ++q) rb += rbfE[(long)grow * 6 + q] * Wrbf6[q * HIDDEN + col];
                v *= rb;
            }
            if (MODE & GM_RES)  v += resid[o];
            if (MODE & GM_OUTF) outF[o] = v;
            if (MODE & GM_OUTB) outB[o] = (__bf16)v;
        }
    }
}

// ---------------------------------------------------------------------------
// Bilinear: msg[idx_ji[t], o] += sum_{b,h} sb[t,b] * x_kj[idx_kj[t], h] * Wbil[o,b,h]
// as 16x1024 @ 1024x128 WMMA per 16-triplet tile; y built in LDS; gathered
// x_kj rows staged via async copies; Wbil K-chunks prefetched.
// ---------------------------------------------------------------------------
__global__ __launch_bounds__(256) void bilinear_kernel(
    const float* __restrict__ cbf, const float* __restrict__ rbfs,
    const int* __restrict__ idx_kj, const int* __restrict__ idx_ji,
    const float* __restrict__ Wsbf,        // 42 x 8
    const __bf16* __restrict__ xkj,        // E x 128
    const __bf16* __restrict__ Wbil,       // [o][b*128+h] bf16 (packed Bt, ldb=1024)
    float* __restrict__ msg, int T) {
    __shared__ __bf16 ldsY[16 * 1024];
    __shared__ __bf16 ldsX[16 * 128];
    __shared__ float  sSb[16][8];
    __shared__ int    sKj[16], sJi[16];

    int t0 = blockIdx.x * 16;
    if (threadIdx.x < 16) {
        int t = t0 + threadIdx.x; if (t >= T) t = T - 1;
        sKj[threadIdx.x] = idx_kj[t];
        sJi[threadIdx.x] = idx_ji[t];
    }
    __syncthreads();

    // Stage the 16 gathered x_kj rows (4KB) with 256 async 16B copies.
    {
        unsigned xbase = (unsigned)(size_t)&ldsX[0];
        int idx = threadIdx.x;
        int r = idx >> 4, ch = idx & 15;
        async_g2l_b128(xbase + (unsigned)idx * 16, xkj + (long)sKj[r] * HIDDEN + ch * 8);
    }

    if (threadIdx.x < 128) {
        int r = threadIdx.x >> 3, b = threadIdx.x & 7;
        int t = t0 + r; if (t >= T) t = T - 1;
        int ekj = sKj[r];
        float s = 0.0f;
        for (int l = 0; l < 7; ++l) {
            float cv = cbf[(long)t * 7 + l];
            for (int q = 0; q < 6; ++q) {
                float sv = rbfs[(long)ekj * 42 + l * 6 + q] * cv;
                s += sv * Wsbf[(l * 6 + q) * 8 + b];
            }
        }
        sSb[r][b] = s;
    }
    wait_async0();
    __syncthreads();

    for (int idx = threadIdx.x; idx < 16 * 1024; idx += 256) {
        int r = idx >> 10;
        int c = idx & 1023;
        int b = c >> 7, h = c & 127;
        float y = sSb[r][b] * (float)ldsX[r * HIDDEN + h];
        ldsY[idx] = (__bf16)y;
    }
    __syncthreads();

    int lane = threadIdx.x & 31;
    int wave = threadIdx.x >> 5;
    int half = lane >> 4;
    int ln   = lane & 15;
    int col  = wave * 16 + ln;

    const __bf16* Bcol = Wbil + (long)col * 1024;
    v8f acc = {};
#pragma unroll 4
    for (int k0 = 0; k0 < 1024; k0 += 32) {
        if (k0 + 256 < 1024) __builtin_prefetch(Bcol + k0 + 256, 0, 0);
        v16bf af, bfv;
        const __bf16* Ar = &ldsY[ln * 1024 + k0];
        const __bf16* Bc = Bcol + k0;
#pragma unroll
        for (int v = 0; v < 8; ++v) {
            int ka = ((v & 4) << 2) + ((v & 3) << 1) + (half << 3);
            af[2 * v]     = Ar[ka];
            af[2 * v + 1] = Ar[ka + 1];
            int kb = ka;
            bfv[2 * v]     = Bc[kb];
            bfv[2 * v + 1] = Bc[kb + 1];
        }
        acc = __builtin_amdgcn_wmma_f32_16x16x32_bf16(false, af, false, bfv,
                                                      (short)0, acc, false, false);
    }

#pragma unroll
    for (int r = 0; r < 8; ++r) {
        int m = r + (half << 3);
        int t = t0 + m;
        if (t >= T) continue;
        atomicAdd(&msg[(long)sJi[m] * HIDDEN + col], acc[r]);
    }
}

// ---------------------------------------------------------------------------
// Host orchestration.
// ---------------------------------------------------------------------------
static void launch_gemm(hipStream_t s, int mode,
                        const __bf16* A, const int* rowmap,
                        const __bf16* Bt, int ldb, int kOff,
                        const float* bias, const float* Cin, const float* resid,
                        const float* rbfE, const float* Wrbf6,
                        float* outF, __bf16* outB, int M) {
    dim3 g((unsigned)((M + 31) / 32));
#define GEMM_CASE(m)                                                            \
    case m:                                                                     \
        gemm_tile_kernel<m><<<g, 256, 0, s>>>(A, rowmap, Bt, ldb, kOff, bias,   \
                                              Cin, resid, rbfE, Wrbf6, outF,    \
                                              outB, M);                         \
        break;
    switch (mode) {
        GEMM_CASE(GM_OUTF)                                            // 32
        GEMM_CASE(GM_CIN | GM_OUTF)                                   // 33
        GEMM_CASE(GM_CIN | GM_BIAS | GM_ACT | GM_OUTF | GM_OUTB)      // 103
        GEMM_CASE(GM_OUTB)                                            // 64
        GEMM_CASE(GM_BIAS | GM_ACT | GM_OUTB)                         // 70
        GEMM_CASE(GM_BIAS | GM_ACT | GM_RES | GM_OUTF)                // 54
        GEMM_CASE(GM_BIAS | GM_ACT | GM_OUTF)                         // 38
        GEMM_CASE(GM_BIAS | GM_ACT | GM_RB | GM_OUTB)                 // 78
        GEMM_CASE(GM_BIAS | GM_ACT | GM_RES | GM_OUTF | GM_OUTB)      // 118
        default: break;
    }
#undef GEMM_CASE
}

extern "C" void kernel_launch(void* const* d_in, const int* in_sizes, int n_in,
                              void* d_out, int out_size, void* d_ws, size_t ws_size,
                              hipStream_t stream) {
    const float* pos        = (const float*)d_in[0];
    const float* emb_table  = (const float*)d_in[1];
    const float* W_rbf_emb  = (const float*)d_in[2];
    const float* b_rbf_emb  = (const float*)d_in[3];
    const float* W_emb      = (const float*)d_in[4];
    const float* b_emb      = (const float*)d_in[5];
    const float* out_Wrbf   = (const float*)d_in[6];
    const float* out_Wdown  = (const float*)d_in[7];
    const float* out_Wl     = (const float*)d_in[8];
    const float* out_bl     = (const float*)d_in[9];
    const float* int_Wrbf   = (const float*)d_in[10];
    const float* int_Wsbf   = (const float*)d_in[11];
    const float* int_Wkj    = (const float*)d_in[12];
    const float* int_bkj    = (const float*)d_in[13];
    const float* int_Wji    = (const float*)d_in[14];
    const float* int_bji    = (const float*)d_in[15];
    const float* int_Wbil   = (const float*)d_in[16];
    const float* int_Wres_b = (const float*)d_in[17];
    const float* int_bres_b = (const float*)d_in[18];
    const float* int_Wlin   = (const float*)d_in[19];
    const float* int_blin   = (const float*)d_in[20];
    const float* int_Wres_a = (const float*)d_in[21];
    const float* int_bres_a = (const float*)d_in[22];
    const int* z      = (const int*)d_in[23];
    const int* edge_i = (const int*)d_in[24];
    const int* edge_j = (const int*)d_in[25];
    const int* idx_i  = (const int*)d_in[26];
    const int* idx_j  = (const int*)d_in[27];
    const int* idx_k  = (const int*)d_in[28];
    const int* idx_kj = (const int*)d_in[29];
    const int* idx_ji = (const int*)d_in[30];

    const int N = in_sizes[23];
    const int E = in_sizes[24];
    const int T = in_sizes[29];
    float* Pout = (float*)d_out;

    char* wp = (char*)d_ws;
    auto alloc = [&](size_t bytes) -> char* {
        char* r = wp;
        wp += (bytes + 255) & ~(size_t)255;
        return r;
    };

    float* zeros  = (float*)alloc(84 * sizeof(float));
    float* normsb = zeros + 42;
    float* rbf    = (float*)alloc((size_t)E * 6 * 4);
    float* rbfs   = (float*)alloc((size_t)E * 42 * 4);
    float* cbf    = (float*)alloc((size_t)T * 7 * 4);
    float* x_f    = (float*)alloc((size_t)E * 128 * 4);
    float* xji    = (float*)alloc((size_t)E * 128 * 4);
    float* msg    = (float*)alloc((size_t)E * 128 * 4);
    float* hm_f   = (float*)alloc((size_t)E * 128 * 4);
    float* tnode  = (float*)alloc((size_t)N * 128 * 4);

    __bf16* h_b     = (__bf16*)alloc((size_t)N * 128 * 2);
    __bf16* rbfh_b  = (__bf16*)alloc((size_t)E * 128 * 2);
    __bf16* x_b     = (__bf16*)alloc((size_t)E * 128 * 2);
    __bf16* xkj_b   = (__bf16*)alloc((size_t)E * 128 * 2);
    __bf16* hm_b    = (__bf16*)alloc((size_t)E * 128 * 2);
    __bf16* u_b     = (__bf16*)alloc((size_t)E * 128 * 2);
    __bf16* tnode_b = (__bf16*)alloc((size_t)N * 128 * 2);
    __bf16* p_b     = (__bf16*)alloc((size_t)N * 128 * 2);
    __bf16* q_b     = (__bf16*)alloc((size_t)N * 128 * 2);

    __bf16* Wemb_t  = (__bf16*)alloc((size_t)128 * 384 * 2);
    __bf16* Wdown_t = (__bf16*)alloc((size_t)7 * 16384 * 2);
    __bf16* Wl_t    = (__bf16*)alloc((size_t)21 * 16384 * 2);
    __bf16* Wji_t   = (__bf16*)alloc((size_t)6 * 16384 * 2);
    __bf16* Wkj_t   = (__bf16*)alloc((size_t)6 * 16384 * 2);
    __bf16* Wlin_t  = (__bf16*)alloc((size_t)6 * 16384 * 2);
    __bf16* Wrb_t   = (__bf16*)alloc((size_t)12 * 16384 * 2);
    __bf16* Wra_t   = (__bf16*)alloc((size_t)24 * 16384 * 2);
    __bf16* Wbil_b  = (__bf16*)alloc((size_t)6 * 131072 * 2);

    const unsigned gE    = (unsigned)((E + 255) / 256);
    const unsigned gT    = (unsigned)((T + 255) / 256);
    const unsigned gE128 = (unsigned)(((size_t)E * 128 + 255) / 256);
    const unsigned gN128 = (unsigned)(((size_t)N * 128 + 255) / 256);

    auto pack = [&](const float* W, __bf16* Bt, int K) {
        int total = 128 * K;
        pack_weight_kernel<<<(total + 255) / 256, 256, 0, stream>>>(W, Bt, K, total);
    };

    hipMemsetAsync(d_out, 0, (size_t)N * 128 * 4, stream);
    bessel_setup_kernel<<<1, 1, 0, stream>>>(zeros, normsb);

    pack(W_emb, Wemb_t, 384);
    for (int i = 0; i < 7; ++i)  pack(out_Wdown  + (size_t)i * 16384, Wdown_t + (size_t)i * 16384, 128);
    for (int i = 0; i < 21; ++i) pack(out_Wl     + (size_t)i * 16384, Wl_t    + (size_t)i * 16384, 128);
    for (int i = 0; i < 6; ++i)  pack(int_Wji    + (size_t)i * 16384, Wji_t   + (size_t)i * 16384, 128);
    for (int i = 0; i < 6; ++i)  pack(int_Wkj    + (size_t)i * 16384, Wkj_t   + (size_t)i * 16384, 128);
    for (int i = 0; i < 6; ++i)  pack(int_Wlin   + (size_t)i * 16384, Wlin_t  + (size_t)i * 16384, 128);
    for (int i = 0; i < 12; ++i) pack(int_Wres_b + (size_t)i * 16384, Wrb_t   + (size_t)i * 16384, 128);
    for (int i = 0; i < 24; ++i) pack(int_Wres_a + (size_t)i * 16384, Wra_t   + (size_t)i * 16384, 128);
    {
        int total = 6 * 131072;  // Wbil already [o][b*128+h]
        f2b_kernel<<<(total + 255) / 256, 256, 0, stream>>>(int_Wbil, Wbil_b, total);
    }

    edge_geom_kernel<<<gE, 256, 0, stream>>>(pos, edge_i, edge_j, zeros, normsb, rbf, rbfs, E);
    triplet_geom_kernel<<<gT, 256, 0, stream>>>(pos, idx_i, idx_j, idx_k, cbf, T);
    gather_emb_kernel<<<gN128, 256, 0, stream>>>(emb_table, z, h_b, N);
    rbf_emb_kernel<<<gE128, 256, 0, stream>>>(rbf, W_rbf_emb, b_rbf_emb, rbfh_b, E);

    // --- embedding ---
    launch_gemm(stream, GM_OUTF, h_b, edge_i, Wemb_t, 384, 0,
                nullptr, nullptr, nullptr, nullptr, nullptr, x_f, nullptr, E);
    launch_gemm(stream, GM_CIN | GM_OUTF, h_b, edge_j, Wemb_t, 384, 128,
                nullptr, x_f, nullptr, nullptr, nullptr, x_f, nullptr, E);
    launch_gemm(stream, GM_CIN | GM_BIAS | GM_ACT | GM_OUTF | GM_OUTB, rbfh_b, nullptr,
                Wemb_t, 384, 256, b_emb, x_f, nullptr, nullptr, nullptr, x_f, x_b, E);

    auto out_block = [&](int k) {
        hipMemsetAsync(tnode, 0, (size_t)N * 128 * 4, stream);
        outblock_scatter_kernel<<<gE128, 256, 0, stream>>>(
            rbf, out_Wrbf + (size_t)k * 768, x_f, edge_i, tnode, E);
        f2b_kernel<<<gN128, 256, 0, stream>>>(tnode, tnode_b, N * 128);
        launch_gemm(stream, GM_OUTB, tnode_b, nullptr, Wdown_t + (size_t)k * 16384, 128, 0,
                    nullptr, nullptr, nullptr, nullptr, nullptr, nullptr, p_b, N);
        launch_gemm(stream, GM_BIAS | GM_ACT | GM_OUTB, p_b, nullptr,
                    Wl_t + (size_t)(k * 3 + 0) * 16384, 128, 0,
                    out_bl + (size_t)(k * 3 + 0) * 128, nullptr, nullptr,
                    nullptr, nullptr, nullptr, q_b, N);
        launch_gemm(stream, GM_BIAS | GM_ACT | GM_OUTB, q_b, nullptr,
                    Wl_t + (size_t)(k * 3 + 1) * 16384, 128, 0,
                    out_bl + (size_t)(k * 3 + 1) * 128, nullptr, nullptr,
                    nullptr, nullptr, nullptr, p_b, N);
        launch_gemm(stream, GM_BIAS | GM_ACT | GM_RES | GM_OUTF, p_b, nullptr,
                    Wl_t + (size_t)(k * 3 + 2) * 16384, 128, 0,
                    out_bl + (size_t)(k * 3 + 2) * 128, nullptr, Pout,
                    nullptr, nullptr, Pout, nullptr, N);
    };

    out_block(0);

    for (int blk = 0; blk < 6; ++blk) {
        launch_gemm(stream, GM_BIAS | GM_ACT | GM_OUTF, x_b, nullptr,
                    Wji_t + (size_t)blk * 16384, 128, 0,
                    int_bji + (size_t)blk * 128, nullptr, nullptr,
                    nullptr, nullptr, xji, nullptr, E);
        launch_gemm(stream, GM_BIAS | GM_ACT | GM_RB | GM_OUTB, x_b, nullptr,
                    Wkj_t + (size_t)blk * 16384, 128, 0,
                    int_bkj + (size_t)blk * 128, nullptr, nullptr,
                    rbf, int_Wrbf + (size_t)blk * 768, nullptr, xkj_b, E);

        hipMemsetAsync(msg, 0, (size_t)E * 128 * 4, stream);
        bilinear_kernel<<<(unsigned)((T + 15) / 16), 256, 0, stream>>>(
            cbf, rbfs, idx_kj, idx_ji, int_Wsbf + (size_t)blk * 336,
            xkj_b, Wbil_b + (size_t)blk * 131072, msg, T);

        add2_kernel<<<gE128, 256, 0, stream>>>(xji, msg, hm_f, hm_b, E * 128);

        launch_gemm(stream, GM_BIAS | GM_ACT | GM_OUTB, hm_b, nullptr,
                    Wrb_t + (size_t)(blk * 2 + 0) * 16384, 128, 0,
                    int_bres_b + (size_t)(blk * 2 + 0) * 128, nullptr, nullptr,
                    nullptr, nullptr, nullptr, u_b, E);
        launch_gemm(stream, GM_BIAS | GM_ACT | GM_RES | GM_OUTF | GM_OUTB, u_b, nullptr,
                    Wrb_t + (size_t)(blk * 2 + 1) * 16384, 128, 0,
                    int_bres_b + (size_t)(blk * 2 + 1) * 128, nullptr, hm_f,
                    nullptr, nullptr, hm_f, hm_b, E);

        launch_gemm(stream, GM_BIAS | GM_ACT | GM_RES | GM_OUTF | GM_OUTB, hm_b, nullptr,
                    Wlin_t + (size_t)blk * 16384, 128, 0,
                    int_blin + (size_t)blk * 128, nullptr, x_f,
                    nullptr, nullptr, x_f, x_b, E);

        for (int r = 0; r < 2; ++r) {
            launch_gemm(stream, GM_BIAS | GM_ACT | GM_OUTB, x_b, nullptr,
                        Wra_t + (size_t)(blk * 4 + 2 * r) * 16384, 128, 0,
                        int_bres_a + (size_t)(blk * 4 + 2 * r) * 128, nullptr, nullptr,
                        nullptr, nullptr, nullptr, u_b, E);
            launch_gemm(stream, GM_BIAS | GM_ACT | GM_RES | GM_OUTF | GM_OUTB, u_b, nullptr,
                        Wra_t + (size_t)(blk * 4 + 2 * r + 1) * 16384, 128, 0,
                        int_bres_a + (size_t)(blk * 4 + 2 * r + 1) * 128, nullptr, x_f,
                        nullptr, nullptr, x_f, x_b, E);
        }

        out_block(blk + 1);
    }

    (void)n_in; (void)out_size; (void)ws_size;
    (void)idx_i; (void)idx_j; (void)idx_k;
}
